// PoolingLayer_51032801411826
// MI455X (gfx1250) — matrix operand order repaired and checked by reading, not compiled, not run on hardware
//
#include <hip/hip_runtime.h>
#include <hip/hip_bf16.h>

typedef __attribute__((ext_vector_type(16))) __bf16 v16bf;
typedef __attribute__((ext_vector_type(8)))  float  v8f;
typedef __attribute__((ext_vector_type(8)))  int    v8i;

#define TPB            256
#define ROWS_PER_BLOCK 128
#define DIM            128
#define K_TOP          32

// ---- LDS layout (bytes) ----
// ylds region double-duty: fp32 x-tile staging (async DMA target) -> y scores
#define Y_PITCH    132                                  // dwords per row (128 + pad)
#define OFF_Y      0
#define SZ_Y       (ROWS_PER_BLOCK * Y_PITCH * 4)       // 67584
#define OFF_VB     (OFF_Y + SZ_Y)                       // 1024 B-fragment records * 32B
#define SZ_VB      32768
#define TOP_PITCH  257                                  // bank step 1 -> conflict-free epilogue
#define OFF_TOPV   OFF_VB                               // top-k phase unions over vB
#define SZ_TOPV    (K_TOP * TOP_PITCH * 4)              // 32896
#define OFF_TOPI   (OFF_TOPV + SZ_TOPV)
#define OFF_RED    (OFF_Y + SZ_Y + 65792)               // after max(union)
#define SMEM_BYTES (OFF_RED + 64)                       // 133440 -> 2 WGs / WGP

static __device__ __forceinline__ unsigned bf16_bits(float f) {
    unsigned u = __float_as_uint(f);
    u += 0x7FFFu + ((u >> 16) & 1u);    // round-to-nearest-even
    return u >> 16;
}
static __device__ __forceinline__ unsigned pack_bf16(float lo, float hi) {
    return bf16_bits(lo) | (bf16_bits(hi) << 16);
}

__global__ void __launch_bounds__(TPB)
topk_gate_wmma_kernel(const float* __restrict__ X,   // [N,128]
                      const float* __restrict__ V,   // [128,128]
                      float* __restrict__ Out,       // [N,32]
                      int N)
{
    extern __shared__ char smem[];
    float*    ylds = (float*)(smem + OFF_Y);
    unsigned* vb   = (unsigned*)(smem + OFF_VB);
    float*    red  = (float*)(smem + OFF_RED);

    const int t    = threadIdx.x;
    const int lane = t & 31;
    const int wave = t >> 5;
    const int rowBlockBase = blockIdx.x * ROWS_PER_BLOCK;

    // ---------- Phase 0a: async-DMA this wave's 16 x-rows (fp32) into its y stripe ----------
    // One GLOBAL_LOAD_ASYNC_TO_LDS_B128 per row: 32 lanes x 16B = 512B = full row,
    // no VGPR round trip; overlaps with the ||v|| / B-fragment prep below (ASYNCcnt).
    {
        const int rowBase = rowBlockBase + wave * 16;
        #pragma unroll
        for (int r = 0; r < 16; ++r) {
            int row = rowBase + r;                                   // wave-uniform branch
            unsigned ldsa = (unsigned)(uintptr_t)(ylds + (wave * 16 + r) * Y_PITCH)
                          + (unsigned)(lane * 16);
            if (row < N) {
                unsigned goff = (unsigned)(row * DIM + lane * 4) * 4u;  // byte offset < 2^31
                asm volatile("global_load_async_to_lds_b128 %0, %1, %2 offset:0"
                             :: "v"(ldsa), "v"(goff), "s"(X) : "memory");
            } else {
                float4 z = make_float4(0.f, 0.f, 0.f, 0.f);
                *(float4*)((char*)smem + (size_t)(wave * 16 + r) * (Y_PITCH * 4) + lane * 16) = z;
            }
        }
    }

    // ---------- Phase 0b: ||v||^2 partials + v -> bf16 B-fragments in LDS ----------
    float ss = 0.f;
    #pragma unroll
    for (int i = 0; i < (DIM * DIM) / TPB; ++i) {
        float f = V[i * TPB + t];
        ss = fmaf(f, f, ss);
    }
    #pragma unroll
    for (int m = 16; m >= 1; m >>= 1) ss += __shfl_xor(ss, m, 32);
    if (lane == 0) red[wave] = ss;

    // B fragment record (nt,ks,lane): lane = khalf*16+n holds 16 consecutive K (bf16)
    #pragma unroll
    for (int i = 0; i < 4; ++i) {
        int rec = i * TPB + t;                 // 0..1023
        int rl = rec & 31, ks = (rec >> 5) & 3, nt = rec >> 7;
        int n = rl & 15, kh = rl >> 4;
        int kbase = ks * 32 + kh * 16;
        unsigned* dst = vb + rec * 8;
        #pragma unroll
        for (int p = 0; p < 8; ++p) {
            float f0 = V[(kbase + 2 * p    ) * DIM + nt * 16 + n];
            float f1 = V[(kbase + 2 * p + 1) * DIM + nt * 16 + n];
            dst[p] = pack_bf16(f0, f1);
        }
    }
    __syncthreads();                            // vB + red visible to all waves

    float inv_len;
    {
        float s = 0.f;
        #pragma unroll
        for (int w = 0; w < 8; ++w) s += red[w];   // deterministic order
        inv_len = 1.0f / sqrtf(s);
    }

    // ---------- Phase 1: wait for DMA, build A fragments (fp32 -> bf16) ----------
    asm volatile("s_wait_asynccnt 0x0" ::: "memory");

    // A fragment (16x32 bf16): lane<16 -> K b..b+15; lane>=16 -> K b+16..b+31
    v16bf afrag[4];
    {
        int m = lane & 15, kh = lane >> 4;
        const float* xrow = ylds + (wave * 16 + m) * Y_PITCH;   // fp32 staging (wave-private)
        #pragma unroll
        for (int ks = 0; ks < 4; ++ks) {
            int b0 = ks * 32 + kh * 16;
            float4 a0 = ((const float4*)(xrow + b0))[0];
            float4 a1 = ((const float4*)(xrow + b0))[1];
            float4 a2 = ((const float4*)(xrow + b0 + 8))[0];
            float4 a3 = ((const float4*)(xrow + b0 + 8))[1];
            v8i ai;
            ai[0] = pack_bf16(a0.x, a0.y); ai[1] = pack_bf16(a0.z, a0.w);
            ai[2] = pack_bf16(a1.x, a1.y); ai[3] = pack_bf16(a1.z, a1.w);
            ai[4] = pack_bf16(a2.x, a2.y); ai[5] = pack_bf16(a2.z, a2.w);
            ai[6] = pack_bf16(a3.x, a3.y); ai[7] = pack_bf16(a3.z, a3.w);
            afrag[ks] = __builtin_bit_cast(v16bf, ai);
        }
    }

    // ---------- Phase 2: y = (x @ v) * inv_len via v_wmma_f32_16x16x32_bf16 ----------
    // Overwrites this wave's own stripe (fragments already in VGPRs; DS ops in-order per wave).
    #pragma unroll
    for (int nt = 0; nt < 8; ++nt) {
        v8f c = {0.f, 0.f, 0.f, 0.f, 0.f, 0.f, 0.f, 0.f};
        #pragma unroll
        for (int ks = 0; ks < 4; ++ks) {
            const int4* bp = (const int4*)(vb + ((nt * 4 + ks) * 32 + lane) * 8);
            int4 b0 = bp[0], b1 = bp[1];
            v8i bi;
            bi[0] = b0.x; bi[1] = b0.y; bi[2] = b0.z; bi[3] = b0.w;
            bi[4] = b1.x; bi[5] = b1.y; bi[6] = b1.z; bi[7] = b1.w;
            v16bf bfrag = __builtin_bit_cast(v16bf, bi);
            c = __builtin_amdgcn_wmma_f32_16x16x32_bf16(
                    false, afrag[ks], false, bfrag, (short)0, c, false, false);
        }
        // C layout: VGPR r, lanes 0-15 -> M=r, lanes 16-31 -> M=r+8; N = lane&15
        int rloc = wave * 16 + 8 * (lane >> 4);
        int col  = nt * 16 + (lane & 15);
        #pragma unroll
        for (int r = 0; r < 8; ++r)
            ylds[(rloc + r) * Y_PITCH + col] = c[r] * inv_len;
    }
    __syncthreads();

    // ---------- Phase 3: lane-per-row streaming top-32 (sorted descending) ----------
    float* topv = (float*)(smem + OFF_TOPV);
    int*   topi = (int*)(smem + OFF_TOPI);
    const int row = rowBlockBase + t;
    if (t < ROWS_PER_BLOCK) {
        #pragma unroll
        for (int s = 0; s < K_TOP; ++s) {
            topv[s * TOP_PITCH + t] = -__builtin_inff();
            topi[s * TOP_PITCH + t] = 0;
        }
        const float4* yr = (const float4*)(ylds + t * Y_PITCH);   // 528B rows: 16B aligned
        float vmin = -__builtin_inff();
        for (int j4 = 0; j4 < DIM / 4; ++j4) {
            float4 f = yr[j4];
            #pragma unroll
            for (int e = 0; e < 4; ++e) {
                float val = (e == 0) ? f.x : (e == 1) ? f.y : (e == 2) ? f.z : f.w;
                if (val > vmin) {                       // early reject vs current 32nd value
                    int s = K_TOP - 1;
                    while (s > 0 && topv[(s - 1) * TOP_PITCH + t] < val) {
                        topv[s * TOP_PITCH + t] = topv[(s - 1) * TOP_PITCH + t];
                        topi[s * TOP_PITCH + t] = topi[(s - 1) * TOP_PITCH + t];
                        --s;
                    }
                    topv[s * TOP_PITCH + t] = val;
                    topi[s * TOP_PITCH + t] = j4 * 4 + e;
                    vmin = topv[(K_TOP - 1) * TOP_PITCH + t];
                }
            }
        }
        // ---------- Phase 4: gate with sigmoid, results back into topv ----------
        if (row < N) {
            #pragma unroll
            for (int s = 0; s < K_TOP; ++s) {
                float yv = topv[s * TOP_PITCH + t];
                int   ix = topi[s * TOP_PITCH + t];
                float xv = X[row * DIM + ix];            // L2-hot (tile just streamed)
                topv[s * TOP_PITCH + t] =
                    xv * __builtin_amdgcn_rcpf(1.0f + __expf(-yv));  // v_rcp_f32, no IEEE div
            }
        }
    }
    __syncthreads();

    // ---------- Phase 5: coalesced store of [128 rows x 32] tile ----------
    const long long outBase = (long long)rowBlockBase * K_TOP;
    #pragma unroll
    for (int i = 0; i < (ROWS_PER_BLOCK * K_TOP) / TPB; ++i) {
        int g  = i * TPB + t;                 // 0..4095
        int rr = g >> 5, s = g & 31;
        if (rowBlockBase + rr < N)
            Out[outBase + g] = topv[s * TOP_PITCH + rr];   // bank step 1: conflict-free
    }
}

extern "C" void kernel_launch(void* const* d_in, const int* in_sizes, int n_in,
                              void* d_out, int out_size, void* d_ws, size_t ws_size,
                              hipStream_t stream) {
    const float* x = (const float*)d_in[0];     // [N,128] fp32
    const float* v = (const float*)d_in[1];     // [128,128] fp32
    // d_in[2] is k==32 (compile-time constant here)
    float* out = (float*)d_out;

    const int N = in_sizes[0] / DIM;            // 500000
    const int grid = (N + ROWS_PER_BLOCK - 1) / ROWS_PER_BLOCK;

    // >64KB dynamic LDS: raise the limit (idempotent, not a stream op)
    hipFuncSetAttribute((const void*)topk_gate_wmma_kernel,
                        hipFuncAttributeMaxDynamicSharedMemorySize, SMEM_BYTES);

    topk_gate_wmma_kernel<<<grid, TPB, SMEM_BYTES, stream>>>(x, v, out, N);
}
